// GAT_24919400251537
// MI455X (gfx1250) — matrix operand order repaired and checked by reading, not compiled, not run on hardware
//
#include <hip/hip_runtime.h>
#include <hip/hip_bf16.h>

typedef __attribute__((ext_vector_type(16))) _Float16 v16h;
typedef __attribute__((ext_vector_type(8)))  float    v8f;
typedef __attribute__((ext_vector_type(4)))  float    v4f;
typedef __attribute__((ext_vector_type(2)))  _Float16 v2h;

#define GAT_ALPHA 0.2f
#define GAT_NEG  -1e12f

// ---------------------------------------------------------------------------
// Phase 1: one pass over distance (268 MB) producing all 5 cij maps.
// Memory-bound: this is the kernel's traffic floor (~11.5us @ 23.3 TB/s).
// ---------------------------------------------------------------------------
__global__ void k_cij(const float* __restrict__ dist,
                      const float* __restrict__ Weh,   // [4][64]
                      const float* __restrict__ We2,   // [64]
                      float* __restrict__ cij, int N) {
    int idx = blockIdx.x * blockDim.x + threadIdx.x;   // i*N + j
    const v4f* d4 = (const v4f*)(dist + (size_t)idx * 64);
    float a0 = 0.f, a1 = 0.f, a2 = 0.f, a3 = 0.f, a4 = 0.f;
#pragma unroll
    for (int q = 0; q < 16; ++q) {
        v4f dv = d4[q];
#pragma unroll
        for (int u = 0; u < 4; ++u) {
            int c = q * 4 + u;
            float x = dv[u];
            a0 += x * Weh[c];
            a1 += x * Weh[64 + c];
            a2 += x * Weh[128 + c];
            a3 += x * Weh[192 + c];
            a4 += x * We2[c];
        }
    }
    size_t NN = (size_t)N * N;
    cij[0 * NN + idx] = a0;
    cij[1 * NN + idx] = a1;
    cij[2 * NN + idx] = a2;
    cij[3 * NN + idx] = a3;
    cij[4 * NN + idx] = a4;
}

// ---------------------------------------------------------------------------
// Layer-1 projection h1 = feats @ Wh[hd].  K=24 far too small for WMMA.
// ---------------------------------------------------------------------------
__global__ void k_h1(const float* __restrict__ feats,  // [B*N][24]
                     const float* __restrict__ Wh,     // [H][24][64]
                     float* __restrict__ h1, int BN) {
    int idx = blockIdx.x * blockDim.x + threadIdx.x;   // hd*BN*64 + r*64 + c
    int c  = idx & 63;
    int r  = (idx >> 6) % BN;
    int hd = idx / (BN * 64);
    const float* f = feats + (size_t)r * 24;
    const float* w = Wh + (size_t)hd * 24 * 64 + c;
    float acc = 0.f;
#pragma unroll
    for (int k = 0; k < 24; ++k) acc += f[k] * w[(size_t)k * 64];
    h1[idx] = acc;
}

// ---------------------------------------------------------------------------
// Per-row attention scalars s_src/s_dst = h@a, plus transposed f16 copy of h
// ([B][64][N]) so the WMMA B operand is K-major (contiguous K pairs).
// ---------------------------------------------------------------------------
__global__ void k_sv(const float* __restrict__ h,   // [B*N][64]
                     const float* __restrict__ a,   // [128]
                     _Float16* __restrict__ hT,     // [B][64][N]
                     float* __restrict__ ss, float* __restrict__ sd,
                     int B, int N) {
    int r = blockIdx.x * blockDim.x + threadIdx.x;
    int b = r / N, n = r % N;
    const float* hr = h + (size_t)r * 64;
    _Float16* ht = hT + ((size_t)b * 64) * N + n;
    float s0 = 0.f, s1 = 0.f;
#pragma unroll
    for (int c = 0; c < 64; ++c) {
        float v = hr[c];
        s0 += v * a[c];
        s1 += v * a[64 + c];
        ht[(size_t)c * N] = (_Float16)v;
    }
    ss[r] = s0;
    sd[r] = s1;
}

// ---------------------------------------------------------------------------
// W2 [256][64] f32 -> W2T [64][256] f16 (K-major for WMMA B operand).
// ---------------------------------------------------------------------------
__global__ void k_convw2(const float* __restrict__ W2, _Float16* __restrict__ W2T) {
    int idx = blockIdx.x * blockDim.x + threadIdx.x;   // k*64 + c
    int c = idx & 63, k = idx >> 6;
    W2T[(size_t)c * 256 + k] = (_Float16)W2[idx];
}

// ---------------------------------------------------------------------------
// Fused GAT attention.  One wave per 16-row tile, 4 waves/block; all waves
// in a block share the same batch b, so the B operand (h tile) is staged
// into LDS ONCE per block via double-buffered global_load_async_to_lds_b128
// (ASYNCcnt pipelined: the next chunk's copies stay in flight during the
// VALU-heavy score/exp pass).  Softmax is 2 sweeps: (1) row max, (2) fused
// unnormalized exp -> WMMA accumulate + row-sum; 1/sum applied at store.
// WMMA layouts per CDNA5 ISA 7.12.2.
// ---------------------------------------------------------------------------
__global__ void __launch_bounds__(128) k_attn(
    const _Float16* __restrict__ hT,   // [B][64][N] f16 (K-major per batch)
    const float* __restrict__ ss,      // [B][N]
    const float* __restrict__ sd,      // [B][N]
    const float* __restrict__ cij,     // [N][N]
    const float* __restrict__ adj,     // [N][N]
    float* __restrict__ outF,          // f32 out or nullptr
    _Float16* __restrict__ outH,       // f16 out or nullptr
    int ostride, int ocol0, int N) {
    __shared__ _Float16 sB[2][64 * 32];   // two 4KB B-tile buffers

    int tid  = threadIdx.x;
    int wave = tid >> 5;
    int lane = tid & 31;
    int tile = blockIdx.x * 4 + wave;
    int tilesPerB = N >> 4;               // 64, divisible by 4 -> uniform b
    int b  = tile / tilesPerB;
    int m0 = (tile % tilesPerB) << 4;
    int row  = lane & 15;
    int half = lane >> 4;
    int m = m0 + row;

    float ssrc = ss[(size_t)b * N + m];
    const float* sdb  = sd  + (size_t)b * N;
    const float* crow = cij + (size_t)m * N;
    const float* arow = adj + (size_t)m * N;
    const _Float16* hTb = hT + (size_t)b * 64 * N;

    // ---- pass 1: row max (lane pair (l, l^16) splits columns even/odd) ----
    float mx = -3.4e38f;
    for (int j = half; j < N; j += 2) {
        float e = ssrc + sdb[j];
        e = e > 0.f ? e : GAT_ALPHA * e;
        e = (arow[j] > 0.f || m == j) ? e : GAT_NEG;
        e += crow[j];
        mx = fmaxf(mx, e);
    }
    mx = fmaxf(mx, __shfl_xor(mx, 16, 32));

    // ---- prologue: async-stage chunk 0 into sB[0] ----
#pragma unroll
    for (int s = tid; s < 256; s += 128) {
        int c = s >> 2, seg = s & 3;
        unsigned lds = (unsigned)(uintptr_t)&sB[0][c * 32 + seg * 8];
        unsigned long long g =
            (unsigned long long)(uintptr_t)(hTb + (size_t)c * N + seg * 8);
        asm volatile("global_load_async_to_lds_b128 %0, %1, off"
                     :: "v"(lds), "v"(g) : "memory");
    }

    // ---- pass 2: fused exp + sum + WMMA accumulate (unnormalized P) ----
    float sum = 0.f;
    v8f acc[4] = {};
    int nch = N >> 5;
    for (int jc = 0; jc < nch; ++jc) {
        int j0 = jc << 5;
        if (jc + 1 < nch) {          // stage next chunk into other buffer
            const _Float16* gb = hTb + (j0 + 32);
            _Float16* sbn = sB[(jc + 1) & 1];
#pragma unroll
            for (int s = tid; s < 256; s += 128) {
                int c = s >> 2, seg = s & 3;
                unsigned lds = (unsigned)(uintptr_t)&sbn[c * 32 + seg * 8];
                unsigned long long g =
                    (unsigned long long)(uintptr_t)(gb + (size_t)c * N + seg * 8);
                asm volatile("global_load_async_to_lds_b128 %0, %1, off"
                             :: "v"(lds), "v"(g) : "memory");
            }
            // allow the 2 newest (next-chunk) copies to remain in flight
            asm volatile("s_wait_asynccnt 0x2" ::: "memory");
        } else {
            asm volatile("s_wait_asynccnt 0x0" ::: "memory");
        }
        __syncthreads();             // current chunk visible to all waves

        // A tile: unnormalized softmax numerators (contiguous K: k = 16*half + i)
        v16h A;
#pragma unroll
        for (int i = 0; i < 16; ++i) {
            int k = (half << 4) + i;
            int j = j0 + k;
            float e = ssrc + sdb[j];
            e = e > 0.f ? e : GAT_ALPHA * e;
            e = (arow[j] > 0.f || m == j) ? e : GAT_NEG;
            e += crow[j];
            float p = __expf(e - mx);
            sum += p;
            A[i] = (_Float16)p;
        }

        // B tiles from LDS (K-major: 32B contiguous per lane -> ds_load_b128)
        const _Float16* sb = sB[jc & 1];
        v16h Bt[4];
#pragma unroll
        for (int t = 0; t < 4; ++t) {
            const _Float16* bc = sb + (t * 16 + row) * 32 + half * 16;
#pragma unroll
            for (int v = 0; v < 8; ++v) {
                v2h p = *(const v2h*)(bc + 2 * v);
                Bt[t][2 * v]     = p.x;
                Bt[t][2 * v + 1] = p.y;
            }
        }
        __syncthreads();             // reads done; buffer may be overwritten

#pragma unroll
        for (int t = 0; t < 4; ++t)
            acc[t] = __builtin_amdgcn_wmma_f32_16x16x32_f16(
                false, A, false, Bt[t], (short)0, acc[t], false, false);
    }

    sum += __shfl_xor(sum, 16, 32);
    float inv = 1.f / sum;

    // ---- store: normalize per row (ds_bpermute shuffle), ReLU ----
    if (outF) {
#pragma unroll
        for (int vv = 0; vv < 8; ++vv) {
            float rinv = __shfl(inv, vv + 8 * half, 32);
            size_t rbase = ((size_t)b * N + m0 + vv + half * 8) * ostride + ocol0;
#pragma unroll
            for (int t = 0; t < 4; ++t)
                outF[rbase + t * 16 + row] = fmaxf(acc[t][vv] * rinv, 0.f);
        }
    }
    if (outH) {
#pragma unroll
        for (int vv = 0; vv < 8; ++vv) {
            float rinv = __shfl(inv, vv + 8 * half, 32);
            size_t rbase = ((size_t)b * N + m0 + vv + half * 8) * ostride + ocol0;
#pragma unroll
            for (int t = 0; t < 4; ++t)
                outH[rbase + t * 16 + row] =
                    (_Float16)fmaxf(acc[t][vv] * rinv, 0.f);
        }
    }
}

// ---------------------------------------------------------------------------
// Layer-2 projection h2 = out1[4096,256] @ W2[256,64], fully WMMA (K=256).
// ---------------------------------------------------------------------------
__global__ void __launch_bounds__(128) k_h2(
    const _Float16* __restrict__ X,   // [R][256] row-major f16
    const _Float16* __restrict__ WT,  // [64][256] f16 (K-major)
    float* __restrict__ h2) {         // [R][64]
    int wave = threadIdx.x >> 5;
    int lane = threadIdx.x & 31;
    int r0   = (blockIdx.x * 4 + wave) << 4;
    int row  = lane & 15;
    int half = lane >> 4;
    const _Float16* xr = X + (size_t)(r0 + row) * 256;
    v8f acc[4] = {};
    for (int j0 = 0; j0 < 256; j0 += 32) {
        v16h A;
#pragma unroll
        for (int v = 0; v < 8; ++v) {
            int k0 = (half << 4) + (v << 1);
            v2h p = *(const v2h*)(xr + j0 + k0);
            A[2 * v]     = p.x;
            A[2 * v + 1] = p.y;
        }
#pragma unroll
        for (int t = 0; t < 4; ++t) {
            const _Float16* wc = WT + (size_t)(t * 16 + row) * 256 + j0 + half * 16;
            __builtin_prefetch(wc + 32, 0, 1);
            v16h Bm;
#pragma unroll
            for (int v = 0; v < 8; ++v) {
                v2h p = *(const v2h*)(wc + 2 * v);
                Bm[2 * v]     = p.x;
                Bm[2 * v + 1] = p.y;
            }
            acc[t] = __builtin_amdgcn_wmma_f32_16x16x32_f16(
                false, A, false, Bm, (short)0, acc[t], false, false);
        }
    }
#pragma unroll
    for (int t = 0; t < 4; ++t)
#pragma unroll
        for (int vv = 0; vv < 8; ++vv)
            h2[((size_t)r0 + vv + half * 8) * 64 + t * 16 + row] = acc[t][vv];
}

// ---------------------------------------------------------------------------
extern "C" void kernel_launch(void* const* d_in, const int* in_sizes, int n_in,
                              void* d_out, int out_size, void* d_ws, size_t ws_size,
                              hipStream_t stream) {
    (void)in_sizes; (void)n_in; (void)out_size; (void)ws_size;
    const float* x    = (const float*)d_in[0];   // [4,1024,12,2] -> feats [B*N][24]
    const float* adj  = (const float*)d_in[1];   // [N][N]
    const float* dist = (const float*)d_in[2];   // [N][N][64]
    const float* Wh   = (const float*)d_in[3];   // [4][24][64]
    const float* ah   = (const float*)d_in[4];   // [4][128]
    const float* Weh  = (const float*)d_in[5];   // [4][64]
    const float* W2   = (const float*)d_in[6];   // [256][64]
    const float* a2   = (const float*)d_in[7];   // [128]
    const float* We2  = (const float*)d_in[8];   // [64]

    const int B = 4, N = 1024, H = 4, C = 64;
    const int BN = B * N;

    char* w = (char*)d_ws;
    float*    cij   = (float*)w;    w += (size_t)5 * N * N * 4;        // 20 MB
    float*    h1    = (float*)w;    w += (size_t)H * BN * C * 4;       // 4 MB
    _Float16* h1T   = (_Float16*)w; w += (size_t)H * B * C * N * 2;    // 2 MB
    float*    ss1   = (float*)w;    w += (size_t)H * BN * 4;
    float*    sd1   = (float*)w;    w += (size_t)H * BN * 4;
    _Float16* out1h = (_Float16*)w; w += (size_t)BN * H * C * 2;       // 2 MB
    _Float16* w2t   = (_Float16*)w; w += (size_t)H * C * C * 2;
    float*    h2    = (float*)w;    w += (size_t)BN * C * 4;
    _Float16* h2T   = (_Float16*)w; w += (size_t)B * C * N * 2;
    float*    ss2   = (float*)w;    w += (size_t)BN * 4;
    float*    sd2   = (float*)w;    w += (size_t)BN * 4;

    // Phase 1: single pass over `distance`, all 5 cij maps.
    k_cij<<<(N * N) / 256, 256, 0, stream>>>(dist, Weh, We2, cij, N);

    // Layer-1 projections + per-row attention scalars (per head).
    k_h1<<<(H * BN * C) / 256, 256, 0, stream>>>(x, Wh, h1, BN);
    for (int hd = 0; hd < H; ++hd)
        k_sv<<<BN / 256, 256, 0, stream>>>(h1 + (size_t)hd * BN * C,
                                           ah + (size_t)hd * 2 * C,
                                           h1T + (size_t)hd * B * C * N,
                                           ss1 + (size_t)hd * BN,
                                           sd1 + (size_t)hd * BN, B, N);

    // Layer-1 fused attention (WMMA + async-LDS staging), relu(concat) f16.
    for (int hd = 0; hd < H; ++hd)
        k_attn<<<(B * (N / 16)) / 4, 128, 0, stream>>>(
            h1T + (size_t)hd * B * C * N,
            ss1 + (size_t)hd * BN, sd1 + (size_t)hd * BN,
            cij + (size_t)hd * N * N, adj,
            nullptr, out1h, H * C, hd * C, N);

    // Layer-2 projection (WMMA, K=256) + scalars.
    k_convw2<<<(H * C * C) / 256, 256, 0, stream>>>(W2, w2t);
    k_h2<<<(BN / 16) / 4, 128, 0, stream>>>(out1h, w2t, h2);
    k_sv<<<BN / 256, 256, 0, stream>>>(h2, a2, h2T, ss2, sd2, B, N);

    // Layer-2 fused attention (WMMA) -> relu -> d_out [B][N][1][64].
    k_attn<<<(B * (N / 16)) / 4, 128, 0, stream>>>(
        h2T, ss2, sd2, cij + (size_t)4 * N * N, adj,
        (float*)d_out, nullptr, C, 0, N);
}
